// SimpleConvGRUCell_40346922778954
// MI455X (gfx1250) — matrix operand order repaired and checked by reading, not compiled, not run on hardware
//
#include <hip/hip_runtime.h>
#include <hip/hip_bf16.h>

// ---------------------------------------------------------------------------
// Types for CDNA5 WMMA (wave32): V_WMMA_F32_16X16X32_BF16
// ---------------------------------------------------------------------------
typedef __bf16 v16bf __attribute__((ext_vector_type(16)));
typedef float  v8f   __attribute__((ext_vector_type(8)));
typedef unsigned int v8u __attribute__((ext_vector_type(8)));
typedef unsigned int u32x4 __attribute__((ext_vector_type(4)));
typedef int i32x4v __attribute__((ext_vector_type(4)));
typedef int i32x8v __attribute__((ext_vector_type(8)));

#define NNODES 20000
#define NEDGES 320000
#define DOUT   256
#define DINH   512   // D_IN + D_OUT

__device__ __forceinline__ v8f vzero8() {
  v8f v;
#pragma unroll
  for (int i = 0; i < 8; ++i) v[i] = 0.0f;
  return v;
}

__device__ __forceinline__ unsigned short f2bf(float f) {
  unsigned int x = __builtin_bit_cast(unsigned int, f);
  x += 0x7FFFu + ((x >> 16) & 1u);   // round-to-nearest-even
  return (unsigned short)(x >> 16);
}

__device__ __forceinline__ unsigned int pkbf(float a, float b) {
  return (unsigned int)f2bf(a) | ((unsigned int)f2bf(b) << 16);
}

__device__ __forceinline__ v8u pack2(uint4 lo, uint4 hi) {
  v8u v;
  v[0] = lo.x; v[1] = lo.y; v[2] = lo.z; v[3] = lo.w;
  v[4] = hi.x; v[5] = hi.y; v[6] = hi.z; v[7] = hi.w;
  return v;
}

// A fragment from bf16 LDS: per-lane 8 bf16 at [m][kbase], 8 at [m][kbase+16]
__device__ __forceinline__ v8u ldA(const unsigned short* p) {
  return pack2(*reinterpret_cast<const uint4*>(p),
               *reinterpret_cast<const uint4*>(p + 16));
}
// A fragment built from f32 LDS tile (TDM-staged), converting to bf16 here
__device__ __forceinline__ v8u ldA_f32(const float* p) {
  float4 a0 = *reinterpret_cast<const float4*>(p);
  float4 a1 = *reinterpret_cast<const float4*>(p + 4);
  float4 b0 = *reinterpret_cast<const float4*>(p + 16);
  float4 b1 = *reinterpret_cast<const float4*>(p + 20);
  v8u v;
  v[0] = pkbf(a0.x, a0.y); v[1] = pkbf(a0.z, a0.w);
  v[2] = pkbf(a1.x, a1.y); v[3] = pkbf(a1.z, a1.w);
  v[4] = pkbf(b0.x, b0.y); v[5] = pkbf(b0.z, b0.w);
  v[6] = pkbf(b1.x, b1.y); v[7] = pkbf(b1.z, b1.w);
  return v;
}
// B fragment: per-lane 8 bf16 at [n][kbase], 8 at [n][kbase+16]
__device__ __forceinline__ v8u ldB(const unsigned short* p) {
  return pack2(*reinterpret_cast<const uint4*>(p),
               *reinterpret_cast<const uint4*>(p + 16));
}

__device__ __forceinline__ v8f wmma_bf16(v8u a, v8u b, v8f c) {
  return __builtin_amdgcn_wmma_f32_16x16x32_bf16(
      false, __builtin_bit_cast(v16bf, a),
      false, __builtin_bit_cast(v16bf, b),
      (short)0, c, false, false);
}

__device__ __forceinline__ float sigm(float x) { return 1.0f / (1.0f + __expf(-x)); }

// ---------------------------------------------------------------------------
// TDM: DMA one 64-row x 32-col f32 tile (row stride 256 elems) into LDS with
// an 8-DWORD pad after every 32 DWORDs (=> padded LDS row stride of 40 f32).
// D# bitfields per CDNA5 ISA. Rows beyond
// rows_valid read as zero (tensor OOB rule), which handles the tail block.
// ---------------------------------------------------------------------------
__device__ __forceinline__ void tdm_load_tile_f32(unsigned lds_byte_off,
                                                  const float* gsrc,
                                                  unsigned rows_valid) {
  unsigned long long ga = (unsigned long long)(const void*)gsrc;
  u32x4 g0;
  g0[0] = 1u;                                        // count=1
  g0[1] = lds_byte_off;                              // lds_addr (bytes)
  g0[2] = (unsigned)(ga & 0xFFFFFFFFu);              // global_addr[31:0]
  g0[3] = (unsigned)((ga >> 32) & 0x1FFFFFFu)        // global_addr[56:32]
        | (2u << 30);                                // type = 2 ("image")
  i32x8v g1;
  g1[0] = (2 << 16)                                  // data_size = 4B
        | (1 << 20)                                  // pad_enable
        | (4 << 22)                                  // pad_interval: 32 DW
        | (7 << 25);                                 // pad_amount: 8 DW
  g1[1] = (int)(32u << 16);                          // tensor_dim0 = 32
  g1[2] = (int)((rows_valid & 0xFFFFu) << 16);       // tensor_dim1[15:0]
  g1[3] = (int)((rows_valid >> 16) & 0xFFFFu)        // tensor_dim1[31:16]
        | (32 << 16);                                // tile_dim0 = 32
  g1[4] = 64;                                        // tile_dim1=64, tile_dim2=0
  g1[5] = 256;                                       // tensor_dim0_stride lo
  g1[6] = 0;                                         // stride hi, dim1_stride lo
  g1[7] = 0;
  i32x4v z4 = {0, 0, 0, 0};
#if __clang_major__ >= 23
  i32x8v z8 = {0, 0, 0, 0, 0, 0, 0, 0};
  __builtin_amdgcn_tensor_load_to_lds(g0, g1, z4, z4, z8, 0);
#else
  __builtin_amdgcn_tensor_load_to_lds(g0, g1, z4, z4, 0);
#endif
}

// ---------------------------------------------------------------------------
// Weight prep: f32 -> bf16 (W_conv transposed so every GEMM's B tile is
// "16 contiguous rows of a row-major [out][K] bf16 matrix").
// ---------------------------------------------------------------------------
__global__ __launch_bounds__(256) void prep_weights_kernel(
    const float* __restrict__ Wfc, const float* __restrict__ Wconv,
    const float* __restrict__ Wih, const float* __restrict__ Whh,
    unsigned short* __restrict__ oFc, unsigned short* __restrict__ oCT,
    unsigned short* __restrict__ oIh, unsigned short* __restrict__ oHh) {
  int i = blockIdx.x * 256 + threadIdx.x;            // grid covers 196608
  if (i < 256 * 512) oFc[i] = f2bf(Wfc[i]);
  if (i < 256 * 256) {
    int nn = i >> 8, kk = i & 255;                   // oCT[n][k] = Wconv[k][n]
    oCT[i] = f2bf(Wconv[kk * 256 + nn]);
  }
  if (i < 768 * 256) { oIh[i] = f2bf(Wih[i]); oHh[i] = f2bf(Whh[i]); }
}

__global__ __launch_bounds__(256) void zero_kernel(float4* __restrict__ p, int n4) {
  int i = blockIdx.x * blockDim.x + threadIdx.x;
  if (i < n4) p[i] = make_float4(0.f, 0.f, 0.f, 0.f);
}

// ---------------------------------------------------------------------------
// Kernel A: xr = relu([x,h] @ W_fc^T + b_fc); m = xr @ W_conv
// Block: 64 rows, 256 threads = 8 waves. Wave w: row-tile (w&3), col-half (w>>2).
// Stage 1 A-tiles are TDM double-buffered (one barrier per K-step).
// ---------------------------------------------------------------------------
#define LDF_STR 40    // 32 + 8-DWORD TDM pad (f32 elements)
#define LXR_STR 264   // 256 + pad (ushorts)

__global__ __launch_bounds__(256) void fc_conv_kernel(
    const float* __restrict__ x, const float* __restrict__ h,
    const unsigned short* __restrict__ Wfc, const float* __restrict__ bfc,
    const unsigned short* __restrict__ WcT,
    float* __restrict__ xr, float* __restrict__ mOut, int nrows) {
  __shared__ float ldsAf[2][64 * LDF_STR];           // TDM-staged f32 tiles
  __shared__ unsigned short ldsXR[64 * LXR_STR];     // xr strip as bf16

  const int tid  = threadIdx.x;
  const int w    = tid >> 5, lane = tid & 31;
  const int n    = lane & 15, hi = lane >> 4;
  const int rt   = w & 3,  ch = w >> 2;
  const int row0 = blockIdx.x * 64;
  const int mrow = rt * 16 + n;
  const unsigned rows_valid = (unsigned)(nrows - row0);
  const bool fullblk = (row0 + 64 <= nrows);

  v8f acc[8];
#pragma unroll
  for (int i = 0; i < 8; ++i) acc[i] = vzero8();

  // ---- stage 1: K = 512 over concat([x, h]), TDM double-buffered ----
  if (tid < 32) {   // wave 0 drives the TDM; EXEC all-ones in this wave
    tdm_load_tile_f32((unsigned)(unsigned long long)(void*)&ldsAf[0][0],
                      &x[(size_t)row0 * 256], rows_valid);
    __builtin_amdgcn_s_wait_tensorcnt(0);
  }
  __syncthreads();

  for (int kb = 0; kb < DINH; kb += 32) {
    const int buf = (kb >> 5) & 1;
    if (tid < 32 && kb + 32 < DINH) {
      int kn = kb + 32;
      const float* src = (kn < 256) ? &x[(size_t)row0 * 256 + kn]
                                    : &h[(size_t)row0 * 256 + (kn - 256)];
      tdm_load_tile_f32((unsigned)(unsigned long long)(void*)&ldsAf[buf ^ 1][0],
                        src, rows_valid);
    }
    v8u afrag = ldA_f32(&ldsAf[buf][mrow * LDF_STR + hi * 8]);
#pragma unroll
    for (int ct = 0; ct < 8; ++ct) {
      int c0 = ch * 128 + ct * 16;
      v8u bfrag = ldB(&Wfc[(size_t)(c0 + n) * DINH + kb + hi * 8]);
      acc[ct] = wmma_bf16(afrag, bfrag, acc[ct]);
    }
    if (tid < 32) __builtin_amdgcn_s_wait_tensorcnt(0);
    __syncthreads();
  }

  // bias + relu; store xr (f32 global + bf16 LDS for stage 2)
#pragma unroll
  for (int ct = 0; ct < 8; ++ct) {
    int col = ch * 128 + ct * 16 + n;
    float bias = bfc[col];
    if (fullblk) {
#pragma unroll
      for (int r = 0; r < 8; ++r) {
        float v = acc[ct][r] + bias; v = v > 0.0f ? v : 0.0f;
        int lrow = rt * 16 + hi * 8 + r;
        xr[(size_t)(row0 + lrow) * 256 + col] = v;
        ldsXR[lrow * LXR_STR + col] = f2bf(v);
      }
    } else {
#pragma unroll
      for (int r = 0; r < 8; ++r) {
        float v = acc[ct][r] + bias; v = v > 0.0f ? v : 0.0f;
        int lrow = rt * 16 + hi * 8 + r;
        if (row0 + lrow < nrows) xr[(size_t)(row0 + lrow) * 256 + col] = v;
        ldsXR[lrow * LXR_STR + col] = f2bf(v);
      }
    }
  }
  __syncthreads();

  // ---- stage 2: m = xr @ W_conv (K = 256, A from LDS) ----
  v8f acc2[8];
#pragma unroll
  for (int i = 0; i < 8; ++i) acc2[i] = vzero8();

  for (int kb = 0; kb < DOUT; kb += 32) {
    v8u afrag = ldA(&ldsXR[mrow * LXR_STR + kb + hi * 8]);
#pragma unroll
    for (int ct = 0; ct < 8; ++ct) {
      int c0 = ch * 128 + ct * 16;
      v8u bfrag = ldB(&WcT[(size_t)(c0 + n) * DOUT + kb + hi * 8]);
      acc2[ct] = wmma_bf16(afrag, bfrag, acc2[ct]);
    }
  }

#pragma unroll
  for (int ct = 0; ct < 8; ++ct) {
    int col = ch * 128 + ct * 16 + n;
    if (fullblk) {
#pragma unroll
      for (int r = 0; r < 8; ++r) {
        int grow = row0 + rt * 16 + hi * 8 + r;
        mOut[(size_t)grow * 256 + col] = acc2[ct][r];
      }
    } else {
#pragma unroll
      for (int r = 0; r < 8; ++r) {
        int grow = row0 + rt * 16 + hi * 8 + r;
        if (grow < nrows) mOut[(size_t)grow * 256 + col] = acc2[ct][r];
      }
    }
  }
}

// ---------------------------------------------------------------------------
// Scatter: agg[dst] += m[src] per edge; m/agg are L2-resident (20.5MB each).
// 4 edges / block, 64 threads (float4) per edge, native f32 atomics.
// ---------------------------------------------------------------------------
__global__ __launch_bounds__(256) void scatter_kernel(
    const float* __restrict__ m, const int* __restrict__ eg,
    float* __restrict__ agg, int nedges) {
  int e = blockIdx.x * 4 + (threadIdx.x >> 6);
  if (e >= nedges) return;
  int c = (threadIdx.x & 63) * 4;
  int s = eg[e];
  int d = eg[nedges + e];
  float4 v = *reinterpret_cast<const float4*>(&m[(size_t)s * 256 + c]);
  float* a = &agg[(size_t)d * 256 + c];
  unsafeAtomicAdd(a + 0, v.x);
  unsafeAtomicAdd(a + 1, v.y);
  unsafeAtomicAdd(a + 2, v.z);
  unsafeAtomicAdd(a + 3, v.w);
}

// ---------------------------------------------------------------------------
// Kernel C: GRU cell. gi = agg@W_ih^T + b_ih ; gh = xr@W_hh^T + b_hh ; gates.
// Block: 32 rows (N % 32 == 0), 8 waves: row-tile (w&1), col-quarter (w>>1),
// 4 col-tiles per wave, 6 accumulators (r/z/n for gi and gh) per tile.
// ---------------------------------------------------------------------------
__global__ __launch_bounds__(256) void gru_kernel(
    const float* __restrict__ agg, const float* __restrict__ xr,
    const unsigned short* __restrict__ Wih, const unsigned short* __restrict__ Whh,
    const float* __restrict__ bih, const float* __restrict__ bhh,
    float* __restrict__ out) {
  __shared__ unsigned short ldsAgg[32 * LXR_STR];
  __shared__ unsigned short ldsXr[32 * LXR_STR];

  const int tid  = threadIdx.x;
  const int row0 = blockIdx.x * 32;

  {  // stage agg and xr strips as bf16
    int r  = tid >> 3;
    int c0 = (tid & 7) * 32;
    size_t g = (size_t)(row0 + r) * 256 + c0;
#pragma unroll
    for (int j = 0; j < 32; j += 4) {
      float4 fa = *reinterpret_cast<const float4*>(&agg[g + j]);
      float4 fx = *reinterpret_cast<const float4*>(&xr[g + j]);
      *reinterpret_cast<uint2*>(&ldsAgg[r * LXR_STR + c0 + j]) =
          make_uint2(pkbf(fa.x, fa.y), pkbf(fa.z, fa.w));
      *reinterpret_cast<uint2*>(&ldsXr[r * LXR_STR + c0 + j]) =
          make_uint2(pkbf(fx.x, fx.y), pkbf(fx.z, fx.w));
    }
  }
  __syncthreads();

  const int w = tid >> 5, lane = tid & 31;
  const int n = lane & 15, hi = lane >> 4;
  const int rt = w & 1, cq = w >> 1;
  const int mrow = rt * 16 + n;

  for (int ct = 0; ct < 4; ++ct) {
    int c0 = cq * 64 + ct * 16;
    v8f aIr = vzero8(), aIz = vzero8(), aIn = vzero8();
    v8f aHr = vzero8(), aHz = vzero8(), aHn = vzero8();

    for (int kb = 0; kb < DOUT; kb += 32) {
      v8u aA = ldA(&ldsAgg[mrow * LXR_STR + kb + hi * 8]);
      v8u aX = ldA(&ldsXr[mrow * LXR_STR + kb + hi * 8]);
      int kk = kb + hi * 8;
      aIr = wmma_bf16(aA, ldB(&Wih[(size_t)(c0 + n) * DOUT + kk]), aIr);
      aIz = wmma_bf16(aA, ldB(&Wih[(size_t)(256 + c0 + n) * DOUT + kk]), aIz);
      aIn = wmma_bf16(aA, ldB(&Wih[(size_t)(512 + c0 + n) * DOUT + kk]), aIn);
      aHr = wmma_bf16(aX, ldB(&Whh[(size_t)(c0 + n) * DOUT + kk]), aHr);
      aHz = wmma_bf16(aX, ldB(&Whh[(size_t)(256 + c0 + n) * DOUT + kk]), aHz);
      aHn = wmma_bf16(aX, ldB(&Whh[(size_t)(512 + c0 + n) * DOUT + kk]), aHn);
    }

    int col = c0 + n;
    float bir = bih[col], biz = bih[256 + col], bin = bih[512 + col];
    float bhr = bhh[col], bhz = bhh[256 + col], bhn = bhh[512 + col];
#pragma unroll
    for (int r = 0; r < 8; ++r) {
      int grow = row0 + rt * 16 + hi * 8 + r;
      float xv = xr[(size_t)grow * 256 + col];
      float rg = sigm(aIr[r] + bir + aHr[r] + bhr);
      float zg = sigm(aIz[r] + biz + aHz[r] + bhz);
      float ng = tanhf(aIn[r] + bin + rg * (aHn[r] + bhn));
      out[(size_t)grow * 256 + col] = (1.0f - zg) * ng + zg * xv;
    }
  }
}

// ---------------------------------------------------------------------------
// Host launcher
// ---------------------------------------------------------------------------
extern "C" void kernel_launch(void* const* d_in, const int* in_sizes, int n_in,
                              void* d_out, int out_size, void* d_ws, size_t ws_size,
                              hipStream_t stream) {
  const float* h_in  = (const float*)d_in[0];
  const float* x_in  = (const float*)d_in[1];
  // d_in[2] = pos (unused by reference)
  const float* W_fc  = (const float*)d_in[3];
  const float* b_fc  = (const float*)d_in[4];
  const float* W_cv  = (const float*)d_in[5];
  const float* W_ih  = (const float*)d_in[6];
  const float* W_hh  = (const float*)d_in[7];
  const float* b_ih  = (const float*)d_in[8];
  const float* b_hh  = (const float*)d_in[9];
  const int*   eg    = (const int*)d_in[10];
  // d_in[11] = edge_index_cand (unused by reference)
  float* out = (float*)d_out;

  char* ws = (char*)d_ws;
  float* xr_f32  = (float*)(ws + 0);                       // 20,480,000 B
  float* m_f32   = (float*)(ws + 20480000);                // 20,480,000 B
  float* agg_f32 = (float*)(ws + 40960000);                // 20,480,000 B
  unsigned short* Wfc_bf = (unsigned short*)(ws + 61440000);   // 262,144 B
  unsigned short* WcT_bf = (unsigned short*)(ws + 61702144);   // 131,072 B
  unsigned short* Wih_bf = (unsigned short*)(ws + 61833216);   // 393,216 B
  unsigned short* Whh_bf = (unsigned short*)(ws + 62226432);   // 393,216 B

  // 1) weights -> bf16 (W_conv transposed)
  prep_weights_kernel<<<768, 256, 0, stream>>>(W_fc, W_cv, W_ih, W_hh,
                                               Wfc_bf, WcT_bf, Wih_bf, Whh_bf);
  // 2) agg = 0
  {
    int n4 = NNODES * DOUT / 4;
    zero_kernel<<<(n4 + 255) / 256, 256, 0, stream>>>((float4*)agg_f32, n4);
  }
  // 3) xr = relu([x,h]W_fc^T+b); m = xr W_conv
  fc_conv_kernel<<<(NNODES + 63) / 64, 256, 0, stream>>>(
      x_in, h_in, Wfc_bf, b_fc, WcT_bf, xr_f32, m_f32, NNODES);
  // 4) agg[dst] += m[src]
  scatter_kernel<<<NEDGES / 4, 256, 0, stream>>>(m_f32, eg, agg_f32, NEDGES);
  // 5) GRU cell -> h_next
  gru_kernel<<<NNODES / 32, 256, 0, stream>>>(
      agg_f32, xr_f32, Wih_bf, Whh_bf, b_ih, b_hh, out);
}
